// ReweightAttentionModule_17729624997973
// MI455X (gfx1250) — matrix operand order repaired and checked by reading, not compiled, not run on hardware
//
#include <hip/hip_runtime.h>

// ---------------- constants ----------------
#define BB   2
#define HH   32
#define SS   1024
#define DD   128
#define HID  4096      // HH*DD
#define LO   512       // HH*R
#define NB   4

typedef __attribute__((ext_vector_type(16))) __bf16 bf16x16;
typedef __attribute__((ext_vector_type(8)))  float  f32x8;

union Frag16 { bf16x16 v; uint4 q[2]; };

#if defined(__has_builtin)
#if __has_builtin(__builtin_amdgcn_global_load_async_to_lds_b128) && \
    __has_builtin(__builtin_amdgcn_s_wait_asynccnt)
#define USE_ASYNC 1
#endif
#endif
#ifndef USE_ASYNC
#define USE_ASYNC 0
#endif

#if USE_ASYNC
typedef int v4i __attribute__((vector_size(16)));
typedef __attribute__((address_space(1))) v4i g1v4i;
typedef __attribute__((address_space(3))) v4i l3v4i;
__device__ __forceinline__ void async_cp_b128(const unsigned short* gsrc, unsigned short* ldst) {
    __builtin_amdgcn_global_load_async_to_lds_b128(
        (g1v4i*)(unsigned long long)gsrc,
        (l3v4i*)(unsigned int)(unsigned long long)ldst,
        0, 0);
}
#endif

__device__ __forceinline__ unsigned short f2bf(float x) {
    unsigned u = __float_as_uint(x);
    unsigned r = ((u >> 16) & 1u) + 0x7FFFu;   // round-to-nearest-even
    return (unsigned short)((u + r) >> 16);
}

__device__ __forceinline__ float sel4(float4 v, int id) {
    return id == 0 ? v.x : id == 1 ? v.y : id == 2 ? v.z : v.w;
}

// ---------------- pack [B,H,S,D] f32 -> [B*S, H*D] bf16 ----------------
__global__ __launch_bounds__(256) void pack_x_kernel(const float* __restrict__ src,
                                                     unsigned short* __restrict__ dst) {
    int t = blockIdx.x * 256 + threadIdx.x;        // t < B*H*S*D = 2^23
    int d = t & 127, s = (t >> 7) & 1023, h = (t >> 17) & 31, b = t >> 22;
    dst[(((size_t)((b << 10) | s)) << 12) | (h << 7) | d] = f2bf(src[t]);
}

// ---------------- elementwise f32 -> bf16 ----------------
__global__ __launch_bounds__(256) void cvt_bf16_kernel(const float* __restrict__ src,
                                                       unsigned short* __restrict__ dst) {
    int t = blockIdx.x * 256 + threadIdx.x;
    dst[t] = f2bf(src[t]);
}

// ---------------- per-(b,k) block id + per-(b,q,n) causal-valid counts ----------------
__global__ __launch_bounds__(256) void prep_counts_kernel(const float* __restrict__ bm,
                                                          float* __restrict__ counts,
                                                          int* __restrict__ blkid) {
    int b = blockIdx.x;
    const float* m = bm + (size_t)b * NB * SS;
    for (int k = threadIdx.x; k < SS; k += 256) {
        int id = 0;
        #pragma unroll
        for (int n = 0; n < NB; ++n) if (m[n * SS + k] > 0.5f) id = n;
        blkid[b * SS + k] = id;
    }
    for (int q = threadIdx.x; q < SS; q += 256) {
        float c0 = 0.f, c1 = 0.f, c2 = 0.f, c3 = 0.f;
        for (int k = 0; k <= q; ++k) {
            c0 += m[k]; c1 += m[SS + k]; c2 += m[2 * SS + k]; c3 += m[3 * SS + k];
        }
        float* cp = counts + ((size_t)b * SS + q) * NB;
        cp[0] = c0; cp[1] = c1; cp[2] = c2; cp[3] = c3;
    }
}

// ---------------- WMMA GEMM: C[M,N] = A[M,K] * W[N,K]^T (bf16 in, bf16 out) ------------
// 32x64 tile per wave (2 A-frags x 4 B-frags -> 8 WMMA per k-step).
// mode 0: Out row-major [M,N].  mode 1: scatter row=(b<<10|s), col=(h<<7|d) -> [B,H,S,D].
__global__ __launch_bounds__(128) void gemm_bf16_kernel(const unsigned short* __restrict__ A,
                                                        const unsigned short* __restrict__ W,
                                                        unsigned short* __restrict__ Out,
                                                        int N, int K, int mode) {
    int wv = threadIdx.x >> 5, lane = threadIdx.x & 31;
    int ln15 = lane & 15, h8 = lane >> 4;
    int ngrps = N >> 6;
    int g = blockIdx.x * 4 + wv;
    int mtile = g / ngrps, ngrp = g % ngrps;

    f32x8 c00 = {0,0,0,0,0,0,0,0};
    f32x8 c01 = c00, c02 = c00, c03 = c00;
    f32x8 c10 = c00, c11 = c00, c12 = c00, c13 = c00;

    const unsigned short* arow0 = A + (size_t)(mtile * 32 + ln15) * K;
    const unsigned short* arow1 = arow0 + (size_t)16 * K;
    const unsigned short* w0 = W + (size_t)(ngrp * 64 +  0 + ln15) * K + h8 * 16;
    const unsigned short* w1 = W + (size_t)(ngrp * 64 + 16 + ln15) * K + h8 * 16;
    const unsigned short* w2 = W + (size_t)(ngrp * 64 + 32 + ln15) * K + h8 * 16;
    const unsigned short* w3 = W + (size_t)(ngrp * 64 + 48 + ln15) * K + h8 * 16;

    for (int kk = 0; kk < K; kk += 32) {
        Frag16 a0, a1;
        a0.q[0] = *(const uint4*)(arow0 + kk + h8 * 16);
        a0.q[1] = *(const uint4*)(arow0 + kk + h8 * 16 + 8);
        a1.q[0] = *(const uint4*)(arow1 + kk + h8 * 16);
        a1.q[1] = *(const uint4*)(arow1 + kk + h8 * 16 + 8);
        Frag16 b0, b1, b2, b3;
        b0.q[0] = ((const uint4*)(w0 + kk))[0]; b0.q[1] = ((const uint4*)(w0 + kk))[1];
        b1.q[0] = ((const uint4*)(w1 + kk))[0]; b1.q[1] = ((const uint4*)(w1 + kk))[1];
        b2.q[0] = ((const uint4*)(w2 + kk))[0]; b2.q[1] = ((const uint4*)(w2 + kk))[1];
        b3.q[0] = ((const uint4*)(w3 + kk))[0]; b3.q[1] = ((const uint4*)(w3 + kk))[1];
        c00 = __builtin_amdgcn_wmma_f32_16x16x32_bf16(false, a0.v, false, b0.v, (short)0, c00, false, false);
        c01 = __builtin_amdgcn_wmma_f32_16x16x32_bf16(false, a0.v, false, b1.v, (short)0, c01, false, false);
        c02 = __builtin_amdgcn_wmma_f32_16x16x32_bf16(false, a0.v, false, b2.v, (short)0, c02, false, false);
        c03 = __builtin_amdgcn_wmma_f32_16x16x32_bf16(false, a0.v, false, b3.v, (short)0, c03, false, false);
        c10 = __builtin_amdgcn_wmma_f32_16x16x32_bf16(false, a1.v, false, b0.v, (short)0, c10, false, false);
        c11 = __builtin_amdgcn_wmma_f32_16x16x32_bf16(false, a1.v, false, b1.v, (short)0, c11, false, false);
        c12 = __builtin_amdgcn_wmma_f32_16x16x32_bf16(false, a1.v, false, b2.v, (short)0, c12, false, false);
        c13 = __builtin_amdgcn_wmma_f32_16x16x32_bf16(false, a1.v, false, b3.v, (short)0, c13, false, false);
    }

    f32x8 cs[2][4] = {{c00, c01, c02, c03}, {c10, c11, c12, c13}};
    #pragma unroll
    for (int i = 0; i < 2; ++i) {
        int rowbase = mtile * 32 + i * 16 + h8 * 8;
        #pragma unroll
        for (int f = 0; f < 4; ++f) {
            int col = ngrp * 64 + f * 16 + ln15;
            #pragma unroll
            for (int r = 0; r < 8; ++r) {
                int orow = rowbase + r;
                unsigned short v = f2bf(cs[i][f][r]);
                if (mode == 0) {
                    Out[(size_t)orow * N + col] = v;
                } else {
                    int bb = orow >> 10, s = orow & 1023;
                    int hh = col >> 7,  d = col & 127;
                    Out[((((size_t)bb * HH + hh) * SS + s) << 7) + d] = v;
                }
            }
        }
    }
}

// ---------------- fused attention / softmax / block-pool / expand ----------------
// grid.x = B*H*(S/64) = 1024 blocks, 128 threads (4 waves); one wave = one 16-query tile.
// K tiles are staged in LDS once per block (async double-buffered when available).
__global__ __launch_bounds__(128) void attn_reweight_kernel(const unsigned short* __restrict__ qp,
                                                            const unsigned short* __restrict__ kp,
                                                            const float* __restrict__ counts,
                                                            const int* __restrict__ blkid_g,
                                                            const float* __restrict__ alpha_p,
                                                            float* __restrict__ out) {
    __shared__ __align__(16) unsigned short ktile[2][16 * DD];   // 2 x 4KB
    __shared__ __align__(16) int    blkid_s[SS];                 // 4KB
    __shared__ __align__(16) float4 bs_s[4][16];                 // 1KB

    int w = threadIdx.x >> 5, lane = threadIdx.x & 31;
    int ln15 = lane & 15, h8 = lane >> 4;
    int blk = blockIdx.x;
    int qchunk = blk & 15, bh = blk >> 4, b = bh >> 5;

    for (int i = threadIdx.x; i < SS; i += 128) blkid_s[i] = blkid_g[b * SS + i];

    int qt = qchunk * 4 + w;
    int qtmax = qchunk * 4 + 3;      // uniform per block
    int qbase = qt * 16;
    int qi = qbase + ln15;

    // B fragments: this lane's query row, contiguous d
    Frag16 bq[4];
    const unsigned short* qrow = qp + ((size_t)bh * SS + qi) * DD;
    #pragma unroll
    for (int f = 0; f < 4; ++f) {
        const uint4* p = (const uint4*)(qrow + f * 32 + h8 * 16);
        bq[f].q[0] = p[0]; bq[f].q[1] = p[1];
    }

    // per-thread K-tile staging slice: row = tid/8, 16 ushorts at dseg = (tid%8)*16
    int srow = threadIdx.x >> 3;
    int dseg = (threadIdx.x & 7) * 16;
    const unsigned short* kbase_g = kp + (size_t)bh * SS * DD;

    const float scaling = 0.08838834764831845f;  // D^-0.5
    float m = -1e30f, l = 0.f;
    float a0 = 0.f, a1 = 0.f, a2 = 0.f, a3 = 0.f;

#if USE_ASYNC
    {   // preload tile 0 into buffer 0 (async, no VGPR roundtrip)
        const unsigned short* gsrc = kbase_g + (size_t)srow * DD + dseg;
        unsigned short* ldst = &ktile[0][srow * DD + dseg];
        async_cp_b128(gsrc, ldst);
        async_cp_b128(gsrc + 8, ldst + 8);
    }
#else
    uint4 p0, p1;
    {
        const unsigned short* gsrc = kbase_g + (size_t)srow * DD + dseg;
        p0 = *(const uint4*)(gsrc);
        p1 = *(const uint4*)(gsrc + 8);
        *(uint4*)&ktile[0][srow * DD + dseg] = p0;
        *(uint4*)&ktile[0][srow * DD + dseg + 8] = p1;
    }
#endif
    __syncthreads();

    for (int kt = 0; kt <= qtmax; ++kt) {
#if USE_ASYNC
        int cur = kt & 1;
        if (kt < qtmax) {   // prefetch next tile into other buffer
            const unsigned short* gsrc = kbase_g + (size_t)((kt + 1) * 16 + srow) * DD + dseg;
            unsigned short* ldst = &ktile[cur ^ 1][srow * DD + dseg];
            async_cp_b128(gsrc, ldst);
            async_cp_b128(gsrc + 8, ldst + 8);
            __builtin_amdgcn_s_wait_asynccnt(2);   // tile kt complete (in-order)
        } else {
            __builtin_amdgcn_s_wait_asynccnt(0);
        }
        __syncthreads();                            // all waves' tile-kt data visible
#else
        int cur = 0;
        if (kt < qtmax) {   // global prefetch of next tile into registers
            const unsigned short* gsrc = kbase_g + (size_t)((kt + 1) * 16 + srow) * DD + dseg;
            p0 = *(const uint4*)(gsrc);
            p1 = *(const uint4*)(gsrc + 8);
        }
#endif
        // score tile: A = 16 keys (from LDS), B = 16 queries (registers)
        f32x8 c = {0,0,0,0,0,0,0,0};
        const unsigned short* kt_l = &ktile[cur][ln15 * DD];
        #pragma unroll
        for (int f = 0; f < 4; ++f) {
            Frag16 a;
            a.q[0] = *(const uint4*)(kt_l + f * 32 + h8 * 16);
            a.q[1] = *(const uint4*)(kt_l + f * 32 + h8 * 16 + 8);
            c = __builtin_amdgcn_wmma_f32_16x16x32_bf16(false, a.v, false, bq[f].v, (short)0, c, false, false);
        }

        int kbase0 = kt * 16 + h8 * 8;
        float s[8];
        float tmax = -1e30f;
        #pragma unroll
        for (int r = 0; r < 8; ++r) {
            int k = kbase0 + r;
            float v = c[r] * scaling;
            s[r] = (k <= qi) ? v : -1e30f;
            tmax = fmaxf(tmax, s[r]);
        }
        if (tmax > -1e29f) {
            if (tmax > m) {
                float sf = __expf(m - tmax);
                l *= sf; a0 *= sf; a1 *= sf; a2 *= sf; a3 *= sf;
                m = tmax;
            }
            #pragma unroll
            for (int r = 0; r < 8; ++r) {
                int k = kbase0 + r;
                if (k <= qi) {
                    float e = __expf(s[r] - m);
                    l += e;
                    int id = blkid_s[k];
                    a0 += (id == 0) ? e : 0.f;
                    a1 += (id == 1) ? e : 0.f;
                    a2 += (id == 2) ? e : 0.f;
                    a3 += (id == 3) ? e : 0.f;
                }
            }
        }
        __syncthreads();    // everyone done reading ktile[cur]
#if !USE_ASYNC
        if (kt < qtmax) {
            *(uint4*)&ktile[0][srow * DD + dseg] = p0;
            *(uint4*)&ktile[0][srow * DD + dseg + 8] = p1;
        }
        __syncthreads();
#endif
    }

    // merge lane pairs (same query, other 8 keys per tile)
    float mo  = __shfl_xor(m, 16, 32);
    float lo  = __shfl_xor(l, 16, 32);
    float a0o = __shfl_xor(a0, 16, 32);
    float a1o = __shfl_xor(a1, 16, 32);
    float a2o = __shfl_xor(a2, 16, 32);
    float a3o = __shfl_xor(a3, 16, 32);
    float M   = fmaxf(m, mo);
    float sfa = __expf(m - M), sfb = __expf(mo - M);
    float L   = l * sfa + lo * sfb;
    a0 = a0 * sfa + a0o * sfb;
    a1 = a1 * sfa + a1o * sfb;
    a2 = a2 * sfa + a2o * sfb;
    a3 = a3 * sfa + a3o * sfb;

    float inv   = 1.f / fmaxf(L, 1e-30f);
    float alpha = alpha_p[0];
    const float* cp = counts + ((size_t)b * SS + qi) * NB;
    float b0 = alpha * a0 * inv / fmaxf(cp[0], 1.f);
    float b1 = alpha * a1 * inv / fmaxf(cp[1], 1.f);
    float b2 = alpha * a2 * inv / fmaxf(cp[2], 1.f);
    float b3 = alpha * a3 * inv / fmaxf(cp[3], 1.f);
    if (h8 == 0) bs_s[w][ln15] = make_float4(b0, b1, b2, b3);
    __syncthreads();

    // expand to full rows: 16 rows x 1024 f32, coalesced b128 stores
    size_t obase = ((size_t)bh * SS + qbase) * SS;
    for (int mr = 0; mr < 16; ++mr) {
        int qq = qbase + mr;
        float4 bsv = bs_s[w][mr];
        #pragma unroll
        for (int it = 0; it < 8; ++it) {
            int k0 = it * 128 + lane * 4;
            int4 id4 = *(const int4*)&blkid_s[k0];
            float4 o;
            o.x = (k0 + 0 <= qq) ? sel4(bsv, id4.x) : 0.f;
            o.y = (k0 + 1 <= qq) ? sel4(bsv, id4.y) : 0.f;
            o.z = (k0 + 2 <= qq) ? sel4(bsv, id4.z) : 0.f;
            o.w = (k0 + 3 <= qq) ? sel4(bsv, id4.w) : 0.f;
            *(float4*)(out + obase + (size_t)mr * SS + k0) = o;
        }
    }
}

// ---------------- host launch ----------------
extern "C" void kernel_launch(void* const* d_in, const int* in_sizes, int n_in,
                              void* d_out, int out_size, void* d_ws, size_t ws_size,
                              hipStream_t stream) {
    const float* q_f   = (const float*)d_in[0];
    const float* k_f   = (const float*)d_in[1];
    const float* Wqa   = (const float*)d_in[2];
    const float* Wqb   = (const float*)d_in[3];
    const float* Wka   = (const float*)d_in[4];
    const float* Wkb   = (const float*)d_in[5];
    const float* alpha = (const float*)d_in[6];
    const float* bm    = (const float*)d_in[7];
    float* out = (float*)d_out;

    char* ws = (char*)d_ws;
    const size_t MB = 1u << 20;
    unsigned short* Xq   = (unsigned short*)(ws);             // 16 MB, reused as qp
    unsigned short* Xk   = (unsigned short*)(ws + 16 * MB);   // 16 MB, reused as kp
    unsigned short* Wqab = (unsigned short*)(ws + 32 * MB);   // 4 MB
    unsigned short* Wqbb = (unsigned short*)(ws + 36 * MB);   // 4 MB
    unsigned short* Wkab = (unsigned short*)(ws + 40 * MB);   // 4 MB
    unsigned short* Wkbb = (unsigned short*)(ws + 44 * MB);   // 4 MB
    unsigned short* Tq   = (unsigned short*)(ws + 48 * MB);   // 2 MB
    unsigned short* Tk   = (unsigned short*)(ws + 50 * MB);   // 2 MB
    float*          cnt  = (float*)(ws + 52 * MB);            // 32 KB
    int*            bid  = (int*)(ws + 52 * MB + 64 * 1024);  // 8 KB

    // pack Q/K -> [B*S, H*D] bf16
    pack_x_kernel<<<32768, 256, 0, stream>>>(q_f, Xq);
    pack_x_kernel<<<32768, 256, 0, stream>>>(k_f, Xk);
    // weights f32 -> bf16 (each 512*4096 = 2M elems)
    cvt_bf16_kernel<<<8192, 256, 0, stream>>>(Wqa, Wqab);
    cvt_bf16_kernel<<<8192, 256, 0, stream>>>(Wqb, Wqbb);
    cvt_bf16_kernel<<<8192, 256, 0, stream>>>(Wka, Wkab);
    cvt_bf16_kernel<<<8192, 256, 0, stream>>>(Wkb, Wkbb);
    // block ids + causal counts
    prep_counts_kernel<<<BB, 256, 0, stream>>>(bm, cnt, bid);

    // GEMM1: T = X @ Wa^T  (M=2048, N=512, K=4096): (2048/32)*(512/64)/4 = 128 blocks
    gemm_bf16_kernel<<<128, 128, 0, stream>>>(Xq, Wqab, Tq, LO, HID, 0);
    gemm_bf16_kernel<<<128, 128, 0, stream>>>(Xk, Wkab, Tk, LO, HID, 0);
    // GEMM2: Y = T @ Wb^T  (M=2048, N=4096, K=512) -> scattered into [B,H,S,D] (reuse X buffers)
    gemm_bf16_kernel<<<1024, 128, 0, stream>>>(Tq, Wqbb, Xq, HID, LO, 1);
    gemm_bf16_kernel<<<1024, 128, 0, stream>>>(Tk, Wkbb, Xk, HID, LO, 1);

    // fused attention + block pooling + expansion
    attn_reweight_kernel<<<BB * HH * (SS / 64), 128, 0, stream>>>(Xq, Xk, cnt, bid, alpha, out);
}
